// DiTBlock_29025388986537
// MI455X (gfx1250) — compile-verified, matrix operand order checked by reading, not run on hardware
//
#include <hip/hip_runtime.h>

// ---------------- problem constants ----------------
#define Hd    1024
#define SIXH  6144
#define Sd    1024
#define Bd    8
#define DHd   64
#define MLPD  4096
#define EPSl  1e-5f
#define MROWS 8192   // B*S

typedef unsigned short u16;
typedef __bf16 bf16;
typedef __attribute__((ext_vector_type(4)))  int    v4i;
typedef __attribute__((ext_vector_type(8)))  __bf16 v8bf;
typedef __attribute__((ext_vector_type(16))) __bf16 v16bf;
typedef __attribute__((ext_vector_type(8)))  float  v8f;

#if defined(__gfx1250__) && \
    __has_builtin(__builtin_amdgcn_global_load_async_to_lds_b128) && \
    __has_builtin(__builtin_amdgcn_s_wait_asynccnt)
#define HAVE_ASYNC 1
#else
#define HAVE_ASYNC 0
#endif

__device__ __forceinline__ u16 f2b(float f) {
  bf16 h = (bf16)f;
  return __builtin_bit_cast(u16, h);
}

// load 16 contiguous bf16 (32B) from LDS as a WMMA fragment register set
__device__ __forceinline__ v16bf ld16(const bf16* p) {
  v8bf a = *(const v8bf*)p;
  v8bf b = *(const v8bf*)(p + 8);
  return __builtin_shufflevector(a, b, 0,1,2,3,4,5,6,7,8,9,10,11,12,13,14,15);
}

// 16B global -> LDS copy; async on gfx1250 when the builtin exists
__device__ __forceinline__ void cp16(bf16* dst, const bf16* src) {
#if HAVE_ASYNC
  __builtin_amdgcn_global_load_async_to_lds_b128(
      (__attribute__((address_space(1))) v4i*)(src),
      (__attribute__((address_space(3))) v4i*)(dst), 0, 0);
#else
  *(v8bf*)dst = *(const v8bf*)src;
#endif
}

__device__ __forceinline__ void wait_async() {
#if HAVE_ASYNC
  __builtin_amdgcn_s_wait_asynccnt(0);
#endif
}

// ---------------- f32 -> bf16 weight cast ----------------
__global__ void k_cast(const float* __restrict__ s, u16* __restrict__ d, int n) {
  for (int i = blockIdx.x * blockDim.x + threadIdx.x; i < n;
       i += gridDim.x * blockDim.x)
    d[i] = f2b(s[i]);
}

// ---------------- c6 = silu(c) @ Wc + bc ----------------
__global__ void __launch_bounds__(256) k_cond(const float* __restrict__ c,
                                              const float* __restrict__ Wc,
                                              const float* __restrict__ bc,
                                              float* __restrict__ c6) {
  __shared__ float cs[Bd * Hd];
  int tid = threadIdx.x;
  for (int i = tid; i < Bd * Hd; i += 256) {
    float v = c[i];
    cs[i] = v / (1.f + __expf(-v));
  }
  __syncthreads();
  int col = blockIdx.x * 256 + tid;
  float acc[Bd];
#pragma unroll
  for (int b = 0; b < Bd; ++b) acc[b] = bc[col];
  for (int k = 0; k < Hd; ++k) {
    float w = Wc[(size_t)k * SIXH + col];
#pragma unroll
    for (int b = 0; b < Bd; ++b) acc[b] += cs[b * Hd + k] * w;
  }
#pragma unroll
  for (int b = 0; b < Bd; ++b) c6[(size_t)b * SIXH + col] = acc[b];
}

// ---------------- LayerNorm + modulate -> bf16 ----------------
__global__ void __launch_bounds__(256) k_ln(const float* __restrict__ x,
                                            const float* __restrict__ c6,
                                            int sh_off, int sc_off,
                                            u16* __restrict__ xm) {
  int row = blockIdx.x, tid = threadIdx.x;
  int b = row >> 10;  // S = 1024 rows per batch
  const float* xr = x + (size_t)row * Hd;
  float4 xv = ((const float4*)xr)[tid];
  float vals[4] = {xv.x, xv.y, xv.z, xv.w};
  float s = vals[0] + vals[1] + vals[2] + vals[3];
  float q = vals[0]*vals[0] + vals[1]*vals[1] + vals[2]*vals[2] + vals[3]*vals[3];
  __shared__ float rs[256], rq[256];
  rs[tid] = s; rq[tid] = q;
  __syncthreads();
  for (int o = 128; o > 0; o >>= 1) {
    if (tid < o) { rs[tid] += rs[tid + o]; rq[tid] += rq[tid + o]; }
    __syncthreads();
  }
  float mean = rs[0] * (1.f / Hd);
  float var  = rq[0] * (1.f / Hd) - mean * mean;
  float rinv = rsqrtf(var + EPSl);
  const float* cb = c6 + (size_t)b * SIXH;
  u16* xo = xm + (size_t)row * Hd;
#pragma unroll
  for (int j = 0; j < 4; ++j) {
    int h = tid * 4 + j;
    float v = (vals[j] - mean) * rinv;
    v = v * (1.f + cb[sc_off + h]) + cb[sh_off + h];
    xo[h] = f2b(v);
  }
}

// ---------------- generic bf16 WMMA GEMM ----------------
// C[M,N] = act((A[M,K] @ B[K,N] + bias) * alpha); optional res + gate*(...)
#define BMt  128
#define BNt  128
#define BKt  64
#define LSTR 80   // padded LDS K-stride (halves); multiple of 16 for 32B frags
#define LBUF (BMt * LSTR)

__global__ void __launch_bounds__(256) k_gemm(
    const u16* __restrict__ Ag_, int lda, const u16* __restrict__ Bg_, int ldb,
    const float* __restrict__ bias, float alpha, int act_gelu,
    const float* __restrict__ resid, const float* __restrict__ c6, int gate_off,
    float* __restrict__ outf, u16* __restrict__ outb, int ldc,
    int M, int N, int K) {
  const bf16* Ag = (const bf16*)Ag_;
  const bf16* Bg = (const bf16*)Bg_;
  __shared__ bf16 lA[2][LBUF];
  __shared__ bf16 lB[2][LBUF];  // stored transposed: [n][k]

  int tid = threadIdx.x, lane = tid & 31, wid = tid >> 5;
  int lrow = lane & 15, lhi = lane >> 4;
  int m0 = blockIdx.y * BMt, n0 = blockIdx.x * BNt;
  int wm0 = (wid & 1) * 64;   // 2 waves over M
  int wn0 = (wid >> 1) * 32;  // 4 waves over N

  v8f acc[4][2] = {};

  // precomputed per-thread staging bases (constant strides, no muls in loop)
  const bf16* gA = Ag + (size_t)(m0 + (tid >> 3)) * lda + ((tid & 7) << 3);
  const bf16* gB = Bg + (size_t)(tid >> 4) * ldb + n0 + ((tid & 15) << 3);
  bf16* sA0 = &lA[0][(tid >> 3) * LSTR + ((tid & 7) << 3)];
  bf16* sB0 = &lB[0][((tid & 15) << 3) * LSTR + (tid >> 4)];
  const size_t dgA = (size_t)32 * lda;  // A: +32 rows per unrolled step
  const size_t dgB = (size_t)16 * ldb;  // B: +16 k-rows per unrolled step
  const size_t dkB = (size_t)BKt * ldb; // B: next K-stage

  auto stage = [&](int kb, int buf) {
    const bf16* a = gA + (size_t)kb * BKt;
    bf16* d = sA0 + buf * LBUF;
#pragma unroll
    for (int it = 0; it < 4; ++it) {   // A tile 128x64, 16B chunks
      cp16(d, a);
      a += dgA;
      d += 32 * LSTR;
    }
    const bf16* b = gB + (size_t)kb * dkB;
    bf16* e = sB0 + buf * LBUF;
#pragma unroll
    for (int it = 0; it < 4; ++it) {   // B tile 64x128, transpose-scatter
      v8bf vv = *(const v8bf*)b;
#pragma unroll
      for (int j = 0; j < 8; ++j) e[j * LSTR] = vv[j];
      b += dgB;
      e += 16;
    }
  };

  int NK = K / BKt;
  stage(0, 0);
  for (int kb = 0; kb < NK; ++kb) {
    wait_async();
    __syncthreads();
    if (kb + 1 < NK) stage(kb + 1, (kb + 1) & 1);
    if (kb + 2 < NK)  // L2 prefetch of a future A tile
      __builtin_prefetch(gA + (size_t)(kb + 2) * BKt, 0, 1);

    int buf = kb & 1;
    const bf16* la = &lA[buf][(wm0 + lrow) * LSTR + (lhi << 4)];
    const bf16* lb = &lB[buf][(wn0 + lrow) * LSTR + (lhi << 4)];
#pragma unroll
    for (int kk = 0; kk < 2; ++kk) {   // two K=32 slices per stage
      v16bf af[4], bfr[2];
#pragma unroll
      for (int mt = 0; mt < 4; ++mt) af[mt] = ld16(la + mt * 16 * LSTR + kk * 32);
#pragma unroll
      for (int nt = 0; nt < 2; ++nt) bfr[nt] = ld16(lb + nt * 16 * LSTR + kk * 32);
#pragma unroll
      for (int mt = 0; mt < 4; ++mt)
#pragma unroll
        for (int nt = 0; nt < 2; ++nt)
          acc[mt][nt] = __builtin_amdgcn_wmma_f32_16x16x32_bf16(
              false, af[mt], false, bfr[nt], (short)0, acc[mt][nt], false, false);
    }
    __syncthreads();
  }

  // epilogue: D layout = VGPR r holds rows r (lanes 0-15) / r+8 (lanes 16-31)
  int halfoff = lhi << 3;
#pragma unroll
  for (int mt = 0; mt < 4; ++mt) {
#pragma unroll
    for (int nt = 0; nt < 2; ++nt) {
      int col = n0 + wn0 + nt * 16 + lrow;
      float bc = bias[col];
#pragma unroll
      for (int r = 0; r < 8; ++r) {
        int row = m0 + wm0 + mt * 16 + halfoff + r;
        float v = (acc[mt][nt][r] + bc) * alpha;
        if (act_gelu) v = 0.5f * v * (1.f + erff(v * 0.70710678118654752f));
        if (resid) {
          float g = c6[(size_t)(row >> 10) * SIXH + gate_off + col];
          v = resid[(size_t)row * ldc + col] + g * v;
        }
        if (outf) outf[(size_t)row * ldc + col] = v;
        else      outb[(size_t)row * ldc + col] = f2b(v);
      }
    }
  }
}

// ---------------- flash attention (per b, head, 64 q rows) ----------------
#define ASTR 80  // LDS stride for 64-wide ch tiles (mult of 16)
#define PSTR 48  // LDS stride for 32-wide key tiles (mult of 16)

__global__ void __launch_bounds__(128) k_attn(const u16* __restrict__ q_,
                                              const u16* __restrict__ k_,
                                              const u16* __restrict__ v_,
                                              u16* __restrict__ y_) {
  const bf16* Q = (const bf16*)q_;
  const bf16* Kp = (const bf16*)k_;
  const bf16* Vp = (const bf16*)v_;
  __shared__ bf16 lQ[64 * ASTR];
  __shared__ bf16 lK[32 * ASTR];
  __shared__ bf16 lVT[64 * PSTR];     // [ch][key]
  __shared__ bf16 lP[4][16 * PSTR];   // per-wave P tile

  int tid = threadIdx.x, lane = tid & 31, wid = tid >> 5;
  int lrow = lane & 15, lhi = lane >> 4;
  int s0 = blockIdx.x * 64, h0 = blockIdx.y * DHd;
  size_t base = (size_t)blockIdx.z * Sd * Hd;

  // per-thread staging bases
  int srow = tid >> 3, scol = (tid & 7) << 3;
  const bf16* gQ = Q + base + (size_t)(s0 + srow) * Hd + h0 + scol;
  const bf16* gK = Kp + base + (size_t)srow * Hd + h0 + scol;
  const bf16* gV = Vp + base + (size_t)srow * Hd + h0 + scol;
  bf16* sQ = &lQ[srow * ASTR + scol];
  bf16* sK = &lK[srow * ASTR + scol];
  bf16* sVT = &lVT[scol * PSTR + srow];

  // stage Q tile 64 x 64 once (4 x 16-row slabs)
#pragma unroll
  for (int it = 0; it < 4; ++it)
    cp16(sQ + it * 16 * ASTR, gQ + (size_t)it * 16 * Hd);
  wait_async();
  __syncthreads();

  v16bf aq[2];
#pragma unroll
  for (int kk = 0; kk < 2; ++kk)
    aq[kk] = ld16(&lQ[(wid * 16 + lrow) * ASTR + kk * 32 + (lhi << 4)]);

  v8f oa[4] = {};
  float mrun[8], lrun[8];
#pragma unroll
  for (int r = 0; r < 8; ++r) { mrun[r] = -1e30f; lrun[r] = 0.f; }

  for (int j = 0; j < Sd / 32; ++j) {
    __syncthreads();  // protect lK / lVT reuse
#pragma unroll
    for (int it = 0; it < 2; ++it)
      cp16(sK + it * 16 * ASTR, gK + (size_t)it * 16 * Hd);
#pragma unroll
    for (int it = 0; it < 2; ++it) {    // V transposed
      v8bf vv = *(const v8bf*)(gV + (size_t)it * 16 * Hd);
#pragma unroll
      for (int jj = 0; jj < 8; ++jj) sVT[jj * PSTR + it * 16] = vv[jj];
    }
    gK += (size_t)32 * Hd;
    gV += (size_t)32 * Hd;
    wait_async();
    __syncthreads();

    // scores: q(16x64) @ k^T(64x32) in two 16-key tiles
    v8f sc[2] = {};
#pragma unroll
    for (int nt = 0; nt < 2; ++nt) {
      const bf16* pk = &lK[(nt * 16 + lrow) * ASTR + (lhi << 4)];
#pragma unroll
      for (int kk = 0; kk < 2; ++kk) {
        v16bf bk = ld16(pk + kk * 32);
        sc[nt] = __builtin_amdgcn_wmma_f32_16x16x32_bf16(
            false, aq[kk], false, bk, (short)0, sc[nt], false, false);
      }
    }

    // online softmax; per-row stats distributed over 16-lane halves
#pragma unroll
    for (int r = 0; r < 8; ++r) {
      float a0 = sc[0][r], a1 = sc[1][r];
      float mx = fmaxf(a0, a1);
      mx = fmaxf(mx, __shfl_xor(mx, 1, 16));
      mx = fmaxf(mx, __shfl_xor(mx, 2, 16));
      mx = fmaxf(mx, __shfl_xor(mx, 4, 16));
      mx = fmaxf(mx, __shfl_xor(mx, 8, 16));
      float mnew = fmaxf(mrun[r], mx);
      float al = __expf(mrun[r] - mnew);
      mrun[r] = mnew;
      a0 = __expf(a0 - mnew);
      a1 = __expf(a1 - mnew);
      float sm = a0 + a1;
      sm += __shfl_xor(sm, 1, 16);
      sm += __shfl_xor(sm, 2, 16);
      sm += __shfl_xor(sm, 4, 16);
      sm += __shfl_xor(sm, 8, 16);
      lrun[r] = lrun[r] * al + sm;
#pragma unroll
      for (int t = 0; t < 4; ++t) oa[t][r] = oa[t][r] * al;
      int prow = r + (lhi << 3);
      lP[wid][prow * PSTR + lrow]      = (bf16)a0;
      lP[wid][prow * PSTR + 16 + lrow] = (bf16)a1;
    }
    __syncthreads();  // make P visible for A-fragment reload

    v16bf ap = ld16(&lP[wid][lrow * PSTR + (lhi << 4)]);
#pragma unroll
    for (int t = 0; t < 4; ++t) {
      v16bf bv = ld16(&lVT[(t * 16 + lrow) * PSTR + (lhi << 4)]);
      oa[t] = __builtin_amdgcn_wmma_f32_16x16x32_bf16(
          false, ap, false, bv, (short)0, oa[t], false, false);
    }
  }

#pragma unroll
  for (int t = 0; t < 4; ++t) {
    int col = h0 + t * 16 + lrow;
#pragma unroll
    for (int r = 0; r < 8; ++r) {
      int row = s0 + wid * 16 + r + (lhi << 3);
      y_[base + (size_t)row * Hd + col] = f2b(oa[t][r] / lrun[r]);
    }
  }
}

// ---------------- host orchestration ----------------
extern "C" void kernel_launch(void* const* d_in, const int* in_sizes, int n_in,
                              void* d_out, int out_size, void* d_ws, size_t ws_size,
                              hipStream_t stream) {
  const float* x   = (const float*)d_in[0];
  const float* c   = (const float*)d_in[1];
  const float* Wc  = (const float*)d_in[2];
  const float* bcv = (const float*)d_in[3];
  const float* Wq  = (const float*)d_in[4];
  const float* bq  = (const float*)d_in[5];
  const float* Wk  = (const float*)d_in[6];
  const float* bk  = (const float*)d_in[7];
  const float* Wv  = (const float*)d_in[8];
  const float* bv  = (const float*)d_in[9];
  const float* Wo  = (const float*)d_in[10];
  const float* bo  = (const float*)d_in[11];
  const float* W1  = (const float*)d_in[12];
  const float* b1  = (const float*)d_in[13];
  const float* W2  = (const float*)d_in[14];
  const float* b2  = (const float*)d_in[15];
  float* outf = (float*)d_out;

  char* p = (char*)d_ws;
  auto alloc = [&](size_t bytes) {
    void* r = (void*)p;
    p += (bytes + 255) & ~(size_t)255;
    return r;
  };
  float* c6 = (float*)alloc((size_t)Bd * SIXH * 4);
  u16* xm  = (u16*)alloc((size_t)MROWS * Hd * 2);
  u16* qb  = (u16*)alloc((size_t)MROWS * Hd * 2);
  u16* kb_ = (u16*)alloc((size_t)MROWS * Hd * 2);
  u16* vb  = (u16*)alloc((size_t)MROWS * Hd * 2);
  u16* yb  = (u16*)alloc((size_t)MROWS * Hd * 2);
  u16* hb  = (u16*)alloc((size_t)MROWS * MLPD * 2);
  u16* Wqb = (u16*)alloc((size_t)Hd * Hd * 2);
  u16* Wkb = (u16*)alloc((size_t)Hd * Hd * 2);
  u16* Wvb = (u16*)alloc((size_t)Hd * Hd * 2);
  u16* Wob = (u16*)alloc((size_t)Hd * Hd * 2);
  u16* W1b = (u16*)alloc((size_t)Hd * MLPD * 2);
  u16* W2b = (u16*)alloc((size_t)MLPD * Hd * 2);

  k_cast<<<512, 256, 0, stream>>>(Wq, Wqb, Hd * Hd);
  k_cast<<<512, 256, 0, stream>>>(Wk, Wkb, Hd * Hd);
  k_cast<<<512, 256, 0, stream>>>(Wv, Wvb, Hd * Hd);
  k_cast<<<512, 256, 0, stream>>>(Wo, Wob, Hd * Hd);
  k_cast<<<2048, 256, 0, stream>>>(W1, W1b, Hd * MLPD);
  k_cast<<<2048, 256, 0, stream>>>(W2, W2b, MLPD * Hd);

  k_cond<<<SIXH / 256, 256, 0, stream>>>(c, Wc, bcv, c6);

  // attention branch
  k_ln<<<MROWS, 256, 0, stream>>>(x, c6, 0, Hd, xm);
  dim3 gH(Hd / BNt, MROWS / BMt);       // (8, 64)
  k_gemm<<<gH, 256, 0, stream>>>(xm, Hd, Wqb, Hd, bq, 1.f / (float)DHd, 0,
                                 nullptr, nullptr, 0, nullptr, qb, Hd,
                                 MROWS, Hd, Hd);
  k_gemm<<<gH, 256, 0, stream>>>(xm, Hd, Wkb, Hd, bk, 1.f, 0,
                                 nullptr, nullptr, 0, nullptr, kb_, Hd,
                                 MROWS, Hd, Hd);
  k_gemm<<<gH, 256, 0, stream>>>(xm, Hd, Wvb, Hd, bv, 1.f, 0,
                                 nullptr, nullptr, 0, nullptr, vb, Hd,
                                 MROWS, Hd, Hd);
  k_attn<<<dim3(Sd / 64, Hd / DHd, Bd), 128, 0, stream>>>(qb, kb_, vb, yb);
  k_gemm<<<gH, 256, 0, stream>>>(yb, Hd, Wob, Hd, bo, 1.f, 0,
                                 x, c6, 2 * Hd, outf, nullptr, Hd,
                                 MROWS, Hd, Hd);

  // MLP branch
  k_ln<<<MROWS, 256, 0, stream>>>(outf, c6, 3 * Hd, 4 * Hd, xm);
  dim3 gM(MLPD / BNt, MROWS / BMt);     // (32, 64)
  k_gemm<<<gM, 256, 0, stream>>>(xm, Hd, W1b, MLPD, b1, 1.f, 1,
                                 nullptr, nullptr, 0, nullptr, hb, MLPD,
                                 MROWS, MLPD, Hd);
  k_gemm<<<gH, 256, 0, stream>>>(hb, MLPD, W2b, Hd, b2, 1.f, 0,
                                 outf, c6, 5 * Hd, outf, nullptr, Hd,
                                 MROWS, Hd, MLPD);
}
